// BestRQModel_6270652252555
// MI455X (gfx1250) — compile-verified, matrix operand order checked by reading, not run on hardware
//
#include <hip/hip_runtime.h>
#include <math.h>

typedef __attribute__((ext_vector_type(16))) _Float16 v16h;
typedef __attribute__((ext_vector_type(8)))  float    v8f;

#define BB 8
#define TT 1200
#define MM 80
#define FRAMES 7
#define STRIDE 4
#define T2 299                 // (1200-7)/4 + 1
#define ROWS (BB*T2)           // 2392
#define ROWS_PAD 2400          // 150 tiles of 16
#define NTILES 150
#define DD 512
#define GG 8
#define DG 64
#define CC 8192
#define NCHUNK 4               // code dim split: 4 x 2048 codes
#define TILES_PER_CHUNK (CC / 16 / NCHUNK)   // 128
#define NPAIRS (TILES_PER_CHUNK / 2)         // 64
#define KSTK (FRAMES*MM)       // 560

#define LAT16_BYTES ((size_t)ROWS_PAD * DD * 2)          // 2,457,600
#define CB16_BYTES  ((size_t)GG * CC * DG * 2)           // 8,388,608
#define PART_N      (NTILES * GG * NCHUNK * 16)          // 76,800 entries
#define PARTV_BYTES ((size_t)PART_N * 4)
#define PARTI_BYTES ((size_t)PART_N * 4)

// ---------------------------------------------------------------------------
// Kernel 0: codebook f32 [C,G,Dg] -> f16 group-major [g][c][Dg]
// ---------------------------------------------------------------------------
__global__ __launch_bounds__(256) void cvt_codebook(const float* __restrict__ cb,
                                                    _Float16* __restrict__ cb16) {
    int i = blockIdx.x * 256 + threadIdx.x;              // over G*C*DG = 4,194,304
    if (i >= GG * CC * DG) return;
    int d = i & (DG - 1);
    int c = (i / DG) & (CC - 1);
    int g = i / (DG * CC);
    cb16[i] = (_Float16)cb[(size_t)c * DD + g * DG + d];
}

// ---------------------------------------------------------------------------
// Kernel 1: stack(7 frames) @ projection, L2-normalize, emit f16 latents.
// 8 rows per block so the 1.1MB projection is reused 8x per L2 read.
// Stacked row = feats[b, 4*t2 : 4*t2+7, :] which is CONTIGUOUS (560 floats).
// ---------------------------------------------------------------------------
__global__ __launch_bounds__(256) void project_norm(const float* __restrict__ feats,
                                                    const float* __restrict__ proj,
                                                    _Float16* __restrict__ lat16) {
    __shared__ float s[8][KSTK];      // 17.9 KB
    __shared__ float red[256];
    __shared__ float tot[8];

    const int tid = threadIdx.x;
    const int rowbase = blockIdx.x * 8;

    for (int idx = tid; idx < 8 * KSTK; idx += 256) {
        int r = idx / KSTK, k = idx - r * KSTK;
        int row = rowbase + r;
        float v = 0.0f;
        if (row < ROWS) {
            int b = row / T2, t2 = row - b * T2;
            v = feats[(size_t)b * TT * MM + (size_t)t2 * STRIDE * MM + k];
        }
        s[r][k] = v;
    }
    __syncthreads();

    float acc[8][2];
#pragma unroll
    for (int r = 0; r < 8; ++r) { acc[r][0] = 0.0f; acc[r][1] = 0.0f; }

    const float* p = proj + tid;
    for (int k = 0; k < KSTK; ++k) {
        float p0 = p[0];
        float p1 = p[256];
        p += DD;
#pragma unroll
        for (int r = 0; r < 8; ++r) {
            float sv = s[r][k];
            acc[r][0] = fmaf(sv, p0, acc[r][0]);
            acc[r][1] = fmaf(sv, p1, acc[r][1]);
        }
    }

#pragma unroll 1
    for (int r = 0; r < 8; ++r) {
        red[tid] = acc[r][0] * acc[r][0] + acc[r][1] * acc[r][1];
        __syncthreads();
        for (int st = 128; st > 0; st >>= 1) {
            if (tid < st) red[tid] += red[tid + st];
            __syncthreads();
        }
        if (tid == 0) tot[r] = red[0];
        __syncthreads();
    }

#pragma unroll
    for (int r = 0; r < 8; ++r) {
        int row = rowbase + r;
        float scale = 1.0f / (sqrtf(tot[r]) + 1e-8f);
        _Float16 v0 = (_Float16)(acc[r][0] * scale);
        _Float16 v1 = (_Float16)(acc[r][1] * scale);
        if (row >= ROWS) { v0 = (_Float16)0.0f; v1 = (_Float16)0.0f; }
        lat16[(size_t)row * DD + tid]       = v0;
        lat16[(size_t)row * DD + tid + 256] = v1;
    }
}

// ---------------------------------------------------------------------------
// search helpers
// ---------------------------------------------------------------------------
__device__ __forceinline__ v8f wmma_pair(v16h a0, v16h a1, v16h b0, v16h b1) {
    v8f c = {};
    c = __builtin_amdgcn_wmma_f32_16x16x32_f16(false, a0, false, b0,
                                               (short)0, c, false, false);
    c = __builtin_amdgcn_wmma_f32_16x16x32_f16(false, a1, false, b1,
                                               (short)0, c, false, false);
    return c;
}

__device__ __forceinline__ void loadb(const _Float16* p, v16h& b0, v16h& b1) {
    b0 = *(const v16h*)(p);
    b1 = *(const v16h*)(p + 32);
}

__device__ __forceinline__ void cmp8(const v8f& c, int n, float* bv, int* bn) {
#pragma unroll
    for (int r = 0; r < 8; ++r) {
        float v = c[r];
        if (v > bv[r]) { bv[r] = v; bn[r] = n; }
    }
}

// ---------------------------------------------------------------------------
// Kernel 2: nearest-code search. One wave per (16-row tile, group, N-chunk).
// Grid (150, 4) x 8 waves = 4800 waves. A fragments (16x64 f16) stay in VGPRs
// for the whole 2048-code chunk. Software-pipelined over 16-code tile PAIRS:
// each body issues WMMAs for pair p, compares pair p-1, issues pair p+1,
// compares pair p -- so every argmax compare is >=4 WMMAs downstream of the
// WMMA that produced its data, covering the WMMA->VALU hazard window without
// v_nops. WGP-scope global_prefetch_b8 (inline asm; the builtin can only
// reach SE/SYS scope) stages the codebook stream L2 -> WGP$.
// ---------------------------------------------------------------------------
__global__ __launch_bounds__(256) void search(const _Float16* __restrict__ lat16,
                                              const _Float16* __restrict__ cb16,
                                              float* __restrict__ partV,
                                              int* __restrict__ partI) {
    const int tile  = blockIdx.x;             // 0..149
    const int chunk = blockIdx.y;             // 0..3
    const int g     = threadIdx.x >> 5;       // one wave per group
    const int lane  = threadIdx.x & 31;
    const int ln    = lane & 15;
    const int hi    = lane >> 4;
    const int rowbase = tile * 16;

    // A fragments: lane ln holds row rowbase+ln; halves split K contiguously (k = 16*hi + s)
    const _Float16* arow = lat16 + (size_t)(rowbase + ln) * DD + g * DG + hi * 16;
    v16h a0 = *(const v16h*)(arow);           // K =  0..31
    v16h a1 = *(const v16h*)(arow + 32);      // K = 32..63

    const int nstart = chunk * (CC / NCHUNK);
    const _Float16* lanebase = cb16 + (size_t)g * CC * DG
                             + (size_t)(nstart + ln) * DG + hi * 16;
    const size_t TSTR = (size_t)16 * DG;      // halves per 16-code tile

    float bvX[8], bvY[8];
    int   bnX[8], bnY[8];
#pragma unroll
    for (int r = 0; r < 8; ++r) {
        bvX[r] = -3.0e38f; bnX[r] = 0;
        bvY[r] = -3.0e38f; bnY[r] = 0;
    }

    v16h b0, b1, b2, b3;

    // ---- prologue: pair 0 (tiles 0,1) -> c2,c3
    loadb(lanebase, b0, b1);
    loadb(lanebase + TSTR, b2, b3);
    v8f c2 = wmma_pair(a0, a1, b0, b1);
    v8f c3 = wmma_pair(a0, a1, b2, b3);
    int n2 = nstart + ln, n3 = n2 + 16;

    int p = 1;
#pragma unroll 1
    for (; p + 1 < NPAIRS; p += 2) {
        // pair p -> c0,c1 ; compare pair p-1 (c2,c3) into set X
        loadb(lanebase + (size_t)(2 * p) * TSTR, b0, b1);
        loadb(lanebase + (size_t)(2 * p + 1) * TSTR, b2, b3);
        unsigned long long pf =
            (unsigned long long)(lanebase + (size_t)(2 * p + 4) * TSTR);
        asm volatile("global_prefetch_b8 %0, off" :: "v"(pf));
        v8f c0 = wmma_pair(a0, a1, b0, b1);
        v8f c1 = wmma_pair(a0, a1, b2, b3);
        int n0 = nstart + 2 * p * 16 + ln, n1 = n0 + 16;
        cmp8(c2, n2, bvX, bnX);
        cmp8(c3, n3, bvX, bnX);
        // pair p+1 -> c2,c3 ; compare pair p (c0,c1) into set Y
        loadb(lanebase + (size_t)(2 * p + 2) * TSTR, b0, b1);
        loadb(lanebase + (size_t)(2 * p + 3) * TSTR, b2, b3);
        c2 = wmma_pair(a0, a1, b0, b1);
        c3 = wmma_pair(a0, a1, b2, b3);
        n2 = n0 + 32; n3 = n0 + 48;
        cmp8(c0, n0, bvY, bnY);
        cmp8(c1, n1, bvY, bnY);
    }
    // ---- epilogue: pair NPAIRS-1, plus drain pair NPAIRS-2
    {
        loadb(lanebase + (size_t)(2 * p) * TSTR, b0, b1);
        loadb(lanebase + (size_t)(2 * p + 1) * TSTR, b2, b3);
        v8f c0 = wmma_pair(a0, a1, b0, b1);
        v8f c1 = wmma_pair(a0, a1, b2, b3);
        int n0 = nstart + 2 * p * 16 + ln, n1 = n0 + 16;
        cmp8(c2, n2, bvX, bnX);
        cmp8(c3, n3, bvX, bnX);
        cmp8(c0, n0, bvY, bnY);
        cmp8(c1, n1, bvY, bnY);
    }

    // fold Y (odd pairs) into X (even pairs): value, lowest-index tie-break
#pragma unroll
    for (int r = 0; r < 8; ++r) {
        if (bvY[r] > bvX[r] || (bvY[r] == bvX[r] && bnY[r] < bnX[r])) {
            bvX[r] = bvY[r]; bnX[r] = bnY[r];
        }
    }

    // argmax across the 16 lanes of each half (lanes 0-15: rows r; 16-31: r+8)
#pragma unroll
    for (int r = 0; r < 8; ++r) {
#pragma unroll
        for (int m = 8; m > 0; m >>= 1) {
            float ov = __shfl_xor(bvX[r], m, 16);
            int   on = __shfl_xor(bnX[r], m, 16);
            if (ov > bvX[r] || (ov == bvX[r] && on < bnX[r])) {
                bvX[r] = ov; bnX[r] = on;
            }
        }
    }

    if (ln == 0) {
        const int base = ((tile * GG + g) * NCHUNK + chunk) * 16;
#pragma unroll
        for (int r = 0; r < 8; ++r) {
            int rl = hi * 8 + r;              // local row 0..15
            partV[base + rl] = bvX[r];
            partI[base + rl] = bnX[r];
        }
    }
}

// ---------------------------------------------------------------------------
// Kernel 2b: fold the 4 N-chunk candidates per (row, group) into final codes.
// ---------------------------------------------------------------------------
__global__ __launch_bounds__(256) void merge_codes(const float* __restrict__ partV,
                                                   const int* __restrict__ partI,
                                                   int* __restrict__ codes) {
    int i = blockIdx.x * 256 + threadIdx.x;   // over ROWS*GG = 19,136
    if (i >= ROWS * GG) return;
    int row = i / GG, g = i - row * GG;
    int tile = row >> 4, rl = row & 15;
    int base = (tile * GG + g) * NCHUNK * 16 + rl;
    float bv = -3.0e38f; int bn = 0;
#pragma unroll
    for (int c = 0; c < NCHUNK; ++c) {
        float v = partV[base + c * 16];
        int   n = partI[base + c * 16];
        if (v > bv || (v == bv && n < bn)) { bv = v; bn = n; }
    }
    codes[i] = bn;
}

// ---------------------------------------------------------------------------
// Kernel 3: gather quantized vectors from the f32 codebook + emit codes.
// d_out = [ quantized: ROWS*512 f32 ][ codes: ROWS*8 (as f32) ]
// ---------------------------------------------------------------------------
__global__ __launch_bounds__(256) void gather_out(const int* __restrict__ codes,
                                                  const float* __restrict__ cb,
                                                  float* __restrict__ out) {
    const int Q = ROWS * DD;
    int i = blockIdx.x * 256 + threadIdx.x;
    if (i < Q) {
        int row  = i >> 9;
        int dcol = i & 511;
        int g = dcol >> 6;
        int d = dcol & 63;
        int idx = codes[row * GG + g];
        out[i] = cb[(size_t)idx * DD + g * DG + d];
    } else if (i < Q + ROWS * GG) {
        out[i] = (float)codes[i - Q];
    }
}

// ---------------------------------------------------------------------------
extern "C" void kernel_launch(void* const* d_in, const int* in_sizes, int n_in,
                              void* d_out, int out_size, void* d_ws, size_t ws_size,
                              hipStream_t stream) {
    const float* feats = (const float*)d_in[0];   // [8,1200,80]
    const float* proj  = (const float*)d_in[1];   // [560,512]
    const float* cb    = (const float*)d_in[2];   // [8192,8,64]

    char* ws = (char*)d_ws;
    size_t off = 0;
    _Float16* lat16 = (_Float16*)(ws + off); off += LAT16_BYTES;
    _Float16* cb16  = (_Float16*)(ws + off); off += CB16_BYTES;
    float*    partV = (float*)(ws + off);    off += PARTV_BYTES;
    int*      partI = (int*)(ws + off);      off += PARTI_BYTES;
    int*      codes = (int*)(ws + off);      off += (size_t)ROWS * GG * 4;
    float*    out   = (float*)d_out;

    cvt_codebook<<<(GG * CC * DG + 255) / 256, 256, 0, stream>>>(cb, cb16);
    project_norm<<<ROWS_PAD / 8, 256, 0, stream>>>(feats, proj, lat16);
    search<<<dim3(NTILES, NCHUNK), 256, 0, stream>>>(lat16, cb16, partV, partI);
    merge_codes<<<(ROWS * GG + 255) / 256, 256, 0, stream>>>(partV, partI, codes);
    gather_out<<<(ROWS * DD + ROWS * GG + 255) / 256, 256, 0, stream>>>(codes, cb, out);
}
